// CML_Attiention_Mask_807453852214
// MI455X (gfx1250) — matrix operand order repaired and checked by reading, not compiled, not run on hardware
//
#include <hip/hip_runtime.h>

// ---------------- constants ----------------
#define NPIX   32768          // B*H*W = 8*64*64
#define CH     256            // DIM
#define HWSZ   4096           // 64*64
#define WIMG   64
#define NHEAD  8
#define DH     32
#define ATT_SCALE 0.17677669529663687f  // 32^-0.5

typedef __attribute__((ext_vector_type(16))) __bf16 v16bf;
typedef __attribute__((ext_vector_type(8)))  float  v8f;

struct __align__(16) u32x4 { unsigned int x, y, z, w; };
struct u32x8 { u32x4 lo, hi; };
struct __align__(16) f32x4 { float x, y, z, w; };

// ---------------- helpers ----------------
__device__ __forceinline__ unsigned short f2bf(float f) {
  unsigned int u = __float_as_uint(f);
  u += 0x7FFFu + ((u >> 16) & 1u);          // round-to-nearest-even
  return (unsigned short)(u >> 16);
}
__device__ __forceinline__ float bf_lo(unsigned int w) { return __uint_as_float(w << 16); }
__device__ __forceinline__ float bf_hi(unsigned int w) { return __uint_as_float(w & 0xFFFF0000u); }

__device__ __forceinline__ void unpack8(const u32x4 w, float* f) {
  f[0] = bf_lo(w.x); f[1] = bf_hi(w.x);
  f[2] = bf_lo(w.y); f[3] = bf_hi(w.y);
  f[4] = bf_lo(w.z); f[5] = bf_hi(w.z);
  f[6] = bf_lo(w.w); f[7] = bf_hi(w.w);
}
__device__ __forceinline__ u32x4 pack8(const float* f) {
  u32x4 o;
  o.x = (unsigned)f2bf(f[0]) | ((unsigned)f2bf(f[1]) << 16);
  o.y = (unsigned)f2bf(f[2]) | ((unsigned)f2bf(f[3]) << 16);
  o.z = (unsigned)f2bf(f[4]) | ((unsigned)f2bf(f[5]) << 16);
  o.w = (unsigned)f2bf(f[6]) | ((unsigned)f2bf(f[7]) << 16);
  return o;
}
__device__ __forceinline__ float sigmoidf_(float x) { return 1.0f / (1.0f + expf(-x)); }
__device__ __forceinline__ float geluf_(float x) {
  return 0.5f * x * (1.0f + erff(x * 0.7071067811865476f));
}

// ---------------- K0: fp32 -> bf16 weight conversion ----------------
__global__ void convb_kernel(const float* __restrict__ src, unsigned short* __restrict__ dst, int n) {
  int i = blockIdx.x * 256 + threadIdx.x;
  if (i < n) dst[i] = f2bf(src[i]);
}

// ---------------- K1: LN1 + transpose BCHW->pixel-major + build mid ----------------
// grid (NPIX/32, 2), block 256.  blockIdx.y: 0=rgb stream, 1=thermal stream.
__global__ __launch_bounds__(256) void prep_kernel(
    const float* __restrict__ in_rgb, const float* __restrict__ in_th,
    const float* __restrict__ g_rgb, const float* __restrict__ b_rgb,
    const float* __restrict__ g_th,  const float* __restrict__ b_th,
    float* __restrict__ xrgb, float* __restrict__ xth,
    unsigned short* __restrict__ rgbn, unsigned short* __restrict__ thn,
    unsigned short* __restrict__ mid) {
  __shared__ float tile[32][257];
  __shared__ float mean_s[32], rstd_s[32];

  const int s  = blockIdx.y;
  const float* in = s ? in_th : in_rgb;
  const float* gm = s ? g_th  : g_rgb;
  const float* bm = s ? b_th  : b_rgb;
  float*          xo = s ? xth : xrgb;
  unsigned short* no = s ? thn : rgbn;

  const int p0  = blockIdx.x * 32;
  const int pxi = threadIdx.x & 31;
  const int csb = threadIdx.x >> 5;     // 0..7
  const int p   = p0 + pxi;
  const int b   = p >> 12;
  const int pix = p & (HWSZ - 1);

  // coalesced gather of 32 pixels x 256 channels
  for (int c = csb; c < CH; c += 8)
    tile[pxi][c] = in[(size_t)b * (CH * HWSZ) + (size_t)c * HWSZ + pix];
  __syncthreads();

  if (threadIdx.x < 32) {
    float sm = 0.f, sq = 0.f;
    for (int c = 0; c < CH; ++c) { float v = tile[threadIdx.x][c]; sm += v; sq += v * v; }
    float m = sm * (1.0f / CH);
    float var = sq * (1.0f / CH) - m * m;
    mean_s[threadIdx.x] = m;
    rstd_s[threadIdx.x] = rsqrtf(var + 1e-5f);
  }
  __syncthreads();

  const int c = threadIdx.x;            // 0..255
  const float g = gm[c], bb = bm[c];
  for (int px = 0; px < 32; ++px) {
    const int pp = p0 + px;
    const float v = tile[px][c];
    xo[(size_t)pp * CH + c] = v;                                   // raw copy (residual)
    const float nv = (v - mean_s[px]) * rstd_s[px] * g + bb;
    const unsigned short h = f2bf(nv);
    no[(size_t)pp * CH + c] = h;                                   // normalized stream
    mid[(size_t)pp * (2 * CH) + s * CH + c] = h;                   // concat for q/gates
  }
}

// ---------------- K2: software-pipelined WMMA bf16 GEMM ----------------
// C[M x Nout] = act(A[M x K] * W[Nout x K]^T + bias) (+resid).
// block 256 = 8 waves; block tile 256(M) x 64(N); wave tile 32(M) x 64(N):
// 2 A fragments x 4 B fragments -> 8 WMMA per K-step, double-buffered loads.
template <int K>
__global__ __launch_bounds__(256) void gemm_bf16_kernel(
    const unsigned short* __restrict__ A, const unsigned short* __restrict__ W,
    const float* __restrict__ bias, const float* __restrict__ resid,
    float* __restrict__ out_f, unsigned short* __restrict__ out_bf,
    int Nout, int act, int scatter) {
  const int lane   = threadIdx.x & 31;
  const int wv     = threadIdx.x >> 5;
  const int l15    = lane & 15;
  const int rowTop = blockIdx.x * 256 + wv * 32;
  const int tileN  = blockIdx.y * 64;
  const int halfA  = (lane < 16) ? 0 : 8;
  const int halfB  = (lane < 16) ? 0 : 8;

  const unsigned short* arow0 = A + (size_t)(rowTop + l15) * K + halfA;
  const unsigned short* arow1 = arow0 + (size_t)16 * K;
  const unsigned short* wrow[4];
#pragma unroll
  for (int cb = 0; cb < 4; ++cb)
    wrow[cb] = W + (size_t)(tileN + cb * 16 + l15) * K + halfB;

  v8f acc[2][4] = {};

  // prologue: K-step 0 fragments
  u32x8 a0, a1, bf[4];
  a0.lo = *(const u32x4*)(arow0);      a0.hi = *(const u32x4*)(arow0 + 16);
  a1.lo = *(const u32x4*)(arow1);      a1.hi = *(const u32x4*)(arow1 + 16);
#pragma unroll
  for (int cb = 0; cb < 4; ++cb) {
    bf[cb].lo = *(const u32x4*)(wrow[cb]);
    bf[cb].hi = *(const u32x4*)(wrow[cb] + 16);
  }

#pragma unroll 2
  for (int k0 = 32; k0 < K; k0 += 32) {
    // issue next K-step loads before consuming current fragments
    u32x8 na0, na1, nb[4];
    na0.lo = *(const u32x4*)(arow0 + k0);  na0.hi = *(const u32x4*)(arow0 + k0 + 16);
    na1.lo = *(const u32x4*)(arow1 + k0);  na1.hi = *(const u32x4*)(arow1 + k0 + 16);
#pragma unroll
    for (int cb = 0; cb < 4; ++cb) {
      nb[cb].lo = *(const u32x4*)(wrow[cb] + k0);
      nb[cb].hi = *(const u32x4*)(wrow[cb] + k0 + 16);
    }
    const v16bf av0 = __builtin_bit_cast(v16bf, a0);
    const v16bf av1 = __builtin_bit_cast(v16bf, a1);
#pragma unroll
    for (int cb = 0; cb < 4; ++cb) {
      const v16bf bv = __builtin_bit_cast(v16bf, bf[cb]);
      acc[0][cb] = __builtin_amdgcn_wmma_f32_16x16x32_bf16(
          false, av0, false, bv, (short)0, acc[0][cb], false, false);
      acc[1][cb] = __builtin_amdgcn_wmma_f32_16x16x32_bf16(
          false, av1, false, bv, (short)0, acc[1][cb], false, false);
    }
    a0 = na0; a1 = na1;
#pragma unroll
    for (int cb = 0; cb < 4; ++cb) bf[cb] = nb[cb];
  }
  {
    const v16bf av0 = __builtin_bit_cast(v16bf, a0);
    const v16bf av1 = __builtin_bit_cast(v16bf, a1);
#pragma unroll
    for (int cb = 0; cb < 4; ++cb) {
      const v16bf bv = __builtin_bit_cast(v16bf, bf[cb]);
      acc[0][cb] = __builtin_amdgcn_wmma_f32_16x16x32_bf16(
          false, av0, false, bv, (short)0, acc[0][cb], false, false);
      acc[1][cb] = __builtin_amdgcn_wmma_f32_16x16x32_bf16(
          false, av1, false, bv, (short)0, acc[1][cb], false, false);
    }
  }

#pragma unroll
  for (int rr = 0; rr < 2; ++rr) {
#pragma unroll
    for (int cb = 0; cb < 4; ++cb) {
      const int col = tileN + cb * 16 + l15;
      const float bv = bias ? bias[col] : 0.0f;
#pragma unroll
      for (int r = 0; r < 8; ++r) {
        const int row = rowTop + rr * 16 + ((lane < 16) ? r : r + 8);
        float v = acc[rr][cb][r] + bv;
        if (act == 1) v = fmaxf(v, 0.0f);
        else if (act == 2) v = geluf_(v);
        if (resid) v += resid[(size_t)row * Nout + col];
        if (out_f) {
          if (scatter) {
            const int bb = row >> 12, pix = row & (HWSZ - 1);
            out_f[(size_t)bb * (CH * HWSZ) + (size_t)col * HWSZ + pix] = v;
          } else {
            out_f[(size_t)row * Nout + col] = v;
          }
        }
        if (out_bf) out_bf[(size_t)row * Nout + col] = f2bf(v);
      }
    }
  }
}

// ---------------- K3: gate head (256 -> 2, sigmoid) ----------------
// wave per pixel; grid NPIX/8, block 256
__global__ __launch_bounds__(256) void gate_kernel(
    const unsigned short* __restrict__ g1, const float* __restrict__ w2,
    const float* __restrict__ b2, float* __restrict__ gates) {
  const int p    = blockIdx.x * 8 + (threadIdx.x >> 5);
  const int lane = threadIdx.x & 31;
  const u32x4 gw = *(const u32x4*)(g1 + (size_t)p * CH + lane * 8);
  float gv[8]; unpack8(gw, gv);
  float s0 = 0.f, s1 = 0.f;
#pragma unroll
  for (int i = 0; i < 8; ++i) {
    const int c = lane * 8 + i;
    s0 += gv[i] * w2[c];
    s1 += gv[i] * w2[CH + c];
  }
#pragma unroll
  for (int off = 16; off >= 1; off >>= 1) {
    s0 += __shfl_xor(s0, off, 32);
    s1 += __shfl_xor(s1, off, 32);
  }
  if (lane == 0) {
    gates[(size_t)p * 2 + 0] = sigmoidf_(s0 + b2[0]);
    gates[(size_t)p * 2 + 1] = sigmoidf_(s1 + b2[1]);
  }
}

// ---------------- K4: 3x3 neighborhood cross-attention + gated mix ----------------
// wave per (pixel, stream); lane = (head, quarter); grid 2*NPIX/8, block 256.
__global__ __launch_bounds__(256) void attn_kernel(
    const unsigned short* __restrict__ qrgb, const unsigned short* __restrict__ qth,
    const unsigned short* __restrict__ kvrgb, const unsigned short* __restrict__ kvth,
    const float* __restrict__ gates,
    const float* __restrict__ xrgb, const float* __restrict__ xth,
    unsigned short* __restrict__ fuse_rgb, unsigned short* __restrict__ fuse_th) {
  const int u    = blockIdx.x * 8 + (threadIdx.x >> 5);
  const int lane = threadIdx.x & 31;
  const int strm = (u >= NPIX) ? 1 : 0;
  const int p    = u - strm * NPIX;

  const unsigned short* q  = strm ? qth   : qrgb;
  const unsigned short* kv = strm ? kvrgb : kvth;   // cross-modal K/V
  const float*          x  = strm ? xth   : xrgb;
  unsigned short*       fo = strm ? fuse_th : fuse_rgb;

  const int b   = p >> 12;
  const int rem = p & (HWSZ - 1);
  const int y   = rem >> 6;
  const int xx  = rem & (WIMG - 1);
  int sy = y - 1;  sy = sy < 0 ? 0 : (sy > WIMG - 3 ? WIMG - 3 : sy);
  int sx = xx - 1; sx = sx < 0 ? 0 : (sx > WIMG - 3 ? WIMG - 3 : sx);

  const int hh  = lane >> 2;          // head 0..7
  const int dof = hh * DH + (lane & 3) * 8;

  float qv[8];
  unpack8(*(const u32x4*)(q + (size_t)p * CH + dof), qv);

  float logit[9];
#pragma unroll
  for (int j = 0; j < 9; ++j) {
    const int pn = (b << 12) + (sy + j / 3) * WIMG + (sx + j % 3);
    float kvv[8];
    unpack8(*(const u32x4*)(kv + (size_t)pn * (2 * CH) + dof), kvv);
    float d = 0.f;
#pragma unroll
    for (int i = 0; i < 8; ++i) d += qv[i] * kvv[i];
    d += __shfl_xor(d, 1, 32);
    d += __shfl_xor(d, 2, 32);        // quad reduce: full 32-dim dot
    logit[j] = -d * ATT_SCALE;        // k negated per reference
  }

  float m = logit[0];
#pragma unroll
  for (int j = 1; j < 9; ++j) m = fmaxf(m, logit[j]);
  float e[9], sum = 0.f;
#pragma unroll
  for (int j = 0; j < 9; ++j) { e[j] = expf(logit[j] - m); sum += e[j]; }
  const float inv = 1.0f / sum;

  float o[8] = {0.f, 0.f, 0.f, 0.f, 0.f, 0.f, 0.f, 0.f};
#pragma unroll
  for (int j = 0; j < 9; ++j) {
    const int pn = (b << 12) + (sy + j / 3) * WIMG + (sx + j % 3);
    float vv[8];
    unpack8(*(const u32x4*)(kv + (size_t)pn * (2 * CH) + CH + dof), vv);
    const float a = e[j] * inv;
#pragma unroll
    for (int i = 0; i < 8; ++i) o[i] += a * vv[i];
  }

  const float g = gates[(size_t)p * 2 + strm];
  float fz[8];
#pragma unroll
  for (int i = 0; i < 8; ++i)
    fz[i] = g * o[i] + (1.0f - g) * x[(size_t)p * CH + dof + i];
  *(u32x4*)(fo + (size_t)p * CH + dof) = pack8(fz);
}

// ---------------- K5: LayerNorm2 (fp32 in, bf16 out) ----------------
// wave per (pixel, stream); grid 2*NPIX/8, block 256
__global__ __launch_bounds__(256) void ln2_kernel(
    const float* __restrict__ res_rgb, const float* __restrict__ res_th,
    const float* __restrict__ g_rgb, const float* __restrict__ b_rgb,
    const float* __restrict__ g_th,  const float* __restrict__ b_th,
    unsigned short* __restrict__ ln_rgb, unsigned short* __restrict__ ln_th) {
  const int u    = blockIdx.x * 8 + (threadIdx.x >> 5);
  const int lane = threadIdx.x & 31;
  const int strm = (u >= NPIX) ? 1 : 0;
  const int p    = u - strm * NPIX;

  const float* r  = strm ? res_th : res_rgb;
  const float* gm = strm ? g_th   : g_rgb;
  const float* bm = strm ? b_th   : b_rgb;
  unsigned short* o = strm ? ln_th : ln_rgb;

  const float* base = r + (size_t)p * CH + lane * 8;
  f32x4 a0 = *(const f32x4*)(base);
  f32x4 a1 = *(const f32x4*)(base + 4);
  float v[8] = {a0.x, a0.y, a0.z, a0.w, a1.x, a1.y, a1.z, a1.w};
  float sm = 0.f, sq = 0.f;
#pragma unroll
  for (int i = 0; i < 8; ++i) { sm += v[i]; sq += v[i] * v[i]; }
#pragma unroll
  for (int off = 16; off >= 1; off >>= 1) {
    sm += __shfl_xor(sm, off, 32);
    sq += __shfl_xor(sq, off, 32);
  }
  const float mean = sm * (1.0f / CH);
  const float var  = sq * (1.0f / CH) - mean * mean;
  const float rstd = rsqrtf(var + 1e-5f);

  float y[8];
#pragma unroll
  for (int i = 0; i < 8; ++i) {
    const int c = lane * 8 + i;
    y[i] = (v[i] - mean) * rstd * gm[c] + bm[c];
  }
  *(u32x4*)(o + (size_t)p * CH + lane * 8) = pack8(y);
}

// ---------------- workspace layout ----------------
static constexpr size_t SZ_F_N256  = (size_t)NPIX * CH * 4;        // 32 MB
static constexpr size_t SZ_BF_N256 = (size_t)NPIX * CH * 2;        // 16 MB
static constexpr size_t SZ_BF_N512 = (size_t)NPIX * 2 * CH * 2;    // 32 MB

// bf16 weight arena element offsets
static constexpr size_t WB_RGB_Q    = 0;
static constexpr size_t WB_TH_Q     = 131072;
static constexpr size_t WB_GATE1    = 262144;
static constexpr size_t WB_RGB_KV   = 393216;
static constexpr size_t WB_TH_KV    = 524288;
static constexpr size_t WB_RGB_PROJ = 655360;
static constexpr size_t WB_TH_PROJ  = 720896;
static constexpr size_t WB_RGB_FC1  = 786432;
static constexpr size_t WB_RGB_FC2  = 917504;
static constexpr size_t WB_TH_FC1   = 1048576;
static constexpr size_t WB_TH_FC2   = 1179648;
static constexpr size_t WB_ELEMS    = 1310720;

static constexpr size_t OFF_WB      = 0;
static constexpr size_t OFF_XRGB    = OFF_WB + WB_ELEMS * 2;       // fp32, aliased later as RES_RGB
static constexpr size_t OFF_XTH     = OFF_XRGB + SZ_F_N256;        // aliased later as RES_TH
static constexpr size_t OFF_MID     = OFF_XTH + SZ_F_N256;         // bf16 Nx512, aliased later as H_RGB
static constexpr size_t OFF_RGBN    = OFF_MID + SZ_BF_N512;        // aliased later as LN2_RGB
static constexpr size_t OFF_THN     = OFF_RGBN + SZ_BF_N256;       // aliased later as LN2_TH
static constexpr size_t OFF_QRGB    = OFF_THN + SZ_BF_N256;        // QRGB..QTH aliased later as H_TH
static constexpr size_t OFF_QTH     = OFF_QRGB + SZ_BF_N256;
static constexpr size_t OFF_KVRGB   = OFF_QTH + SZ_BF_N256;        // bf16 Nx512 (k|v)
static constexpr size_t OFF_KVTH    = OFF_KVRGB + SZ_BF_N512;
static constexpr size_t OFF_GATES   = OFF_KVTH + SZ_BF_N512;       // fp32 Nx2
static constexpr size_t OFF_FUSERGB = OFF_GATES + (size_t)NPIX * 2 * 4;  // also used first as G1
static constexpr size_t OFF_FUSETH  = OFF_FUSERGB + SZ_BF_N256;

// input indices (setup_inputs dict order, params flattened in insertion order)
enum {
  IN_RGB = 0, IN_TH = 1,
  P_RGB_N1_G = 2, P_RGB_N1_B = 3, P_TH_N1_G = 4, P_TH_N1_B = 5,
  P_RGB_N2_G = 6, P_RGB_N2_B = 7, P_TH_N2_G = 8, P_TH_N2_B = 9,
  P_RGB_Q_W = 10, P_RGB_Q_B = 11, P_TH_Q_W = 12, P_TH_Q_B = 13,
  P_RGB_KV_W = 14, P_RGB_KV_B = 15, P_TH_KV_W = 16, P_TH_KV_B = 17,
  P_RGB_PROJ_W = 18, P_RGB_PROJ_B = 19, P_TH_PROJ_W = 20, P_TH_PROJ_B = 21,
  P_GATE_W1 = 22, P_GATE_B1 = 23, P_GATE_W2 = 24, P_GATE_B2 = 25,
  P_RGB_FC1 = 26, P_RGB_FC2 = 27, P_TH_FC1 = 28, P_TH_FC2 = 29
};

extern "C" void kernel_launch(void* const* d_in, const int* in_sizes, int n_in,
                              void* d_out, int out_size, void* d_ws, size_t ws_size,
                              hipStream_t stream) {
  (void)in_sizes; (void)n_in; (void)out_size; (void)ws_size;
  char* ws = (char*)d_ws;
  unsigned short* WB      = (unsigned short*)(ws + OFF_WB);
  float*          XRGB    = (float*)(ws + OFF_XRGB);
  float*          XTH     = (float*)(ws + OFF_XTH);
  unsigned short* MID     = (unsigned short*)(ws + OFF_MID);
  unsigned short* RGBN    = (unsigned short*)(ws + OFF_RGBN);
  unsigned short* THN     = (unsigned short*)(ws + OFF_THN);
  unsigned short* QRGB    = (unsigned short*)(ws + OFF_QRGB);
  unsigned short* QTH     = (unsigned short*)(ws + OFF_QTH);
  unsigned short* KVRGB   = (unsigned short*)(ws + OFF_KVRGB);
  unsigned short* KVTH    = (unsigned short*)(ws + OFF_KVTH);
  float*          GATES   = (float*)(ws + OFF_GATES);
  unsigned short* FUSERGB = (unsigned short*)(ws + OFF_FUSERGB);
  unsigned short* FUSETH  = (unsigned short*)(ws + OFF_FUSETH);
  // aliases (sequentially safe)
  unsigned short* G1      = FUSERGB;                  // gate hidden (before fuse written)
  float*          RES_RGB = XRGB;                     // residual stream overwrites raw x
  float*          RES_TH  = XTH;
  unsigned short* LN2_RGB = RGBN;                     // free after kv GEMMs
  unsigned short* LN2_TH  = THN;
  unsigned short* H_RGB   = MID;                      // free after q/gate GEMMs
  unsigned short* H_TH    = QRGB;                     // QRGB+QTH span = 32MB, free after attention

  auto f32in = [&](int i) { return (const float*)d_in[i]; };

  // ---- K0: weight conversion to bf16 ----
  struct WItem { int idx; size_t off; int n; };
  const WItem items[11] = {
    {P_RGB_Q_W,    WB_RGB_Q,    131072}, {P_TH_Q_W,  WB_TH_Q,   131072},
    {P_GATE_W1,    WB_GATE1,    131072},
    {P_RGB_KV_W,   WB_RGB_KV,   131072}, {P_TH_KV_W, WB_TH_KV,  131072},
    {P_RGB_PROJ_W, WB_RGB_PROJ,  65536}, {P_TH_PROJ_W, WB_TH_PROJ, 65536},
    {P_RGB_FC1,    WB_RGB_FC1,  131072}, {P_RGB_FC2, WB_RGB_FC2, 131072},
    {P_TH_FC1,     WB_TH_FC1,   131072}, {P_TH_FC2,  WB_TH_FC2,  131072}};
  for (int i = 0; i < 11; ++i)
    convb_kernel<<<dim3((items[i].n + 255) / 256), dim3(256), 0, stream>>>(
        f32in(items[i].idx), WB + items[i].off, items[i].n);

  // ---- K1: LN1 + transpose + mid ----
  prep_kernel<<<dim3(NPIX / 32, 2), dim3(256), 0, stream>>>(
      f32in(IN_RGB), f32in(IN_TH),
      f32in(P_RGB_N1_G), f32in(P_RGB_N1_B), f32in(P_TH_N1_G), f32in(P_TH_N1_B),
      XRGB, XTH, RGBN, THN, MID);

  const dim3 blk(256);
  const dim3 g4(NPIX / 256, 4);   // Nout=256, block tile 256x64
  const dim3 g8(NPIX / 256, 8);   // Nout=512

  // ---- q / gate1 / kv GEMMs ----
  gemm_bf16_kernel<512><<<g4, blk, 0, stream>>>(MID, WB + WB_RGB_Q, f32in(P_RGB_Q_B),
      nullptr, nullptr, QRGB, 256, 0, 0);
  gemm_bf16_kernel<512><<<g4, blk, 0, stream>>>(MID, WB + WB_TH_Q, f32in(P_TH_Q_B),
      nullptr, nullptr, QTH, 256, 0, 0);
  gemm_bf16_kernel<512><<<g4, blk, 0, stream>>>(MID, WB + WB_GATE1, f32in(P_GATE_B1),
      nullptr, nullptr, G1, 256, /*relu*/1, 0);
  gemm_bf16_kernel<256><<<g8, blk, 0, stream>>>(RGBN, WB + WB_RGB_KV, f32in(P_RGB_KV_B),
      nullptr, nullptr, KVRGB, 512, 0, 0);
  gemm_bf16_kernel<256><<<g8, blk, 0, stream>>>(THN, WB + WB_TH_KV, f32in(P_TH_KV_B),
      nullptr, nullptr, KVTH, 512, 0, 0);

  // ---- gate head (reads G1 before fuse buffers are overwritten) ----
  gate_kernel<<<dim3(NPIX / 8), blk, 0, stream>>>(G1, f32in(P_GATE_W2),
      f32in(P_GATE_B2), GATES);

  // ---- neighborhood cross-attention + gated mix ----
  attn_kernel<<<dim3(2 * NPIX / 8), blk, 0, stream>>>(
      QRGB, QTH, KVRGB, KVTH, GATES, XRGB, XTH, FUSERGB, FUSETH);

  // ---- proj GEMMs with residual (write fp32 residual stream over X) ----
  gemm_bf16_kernel<256><<<g4, blk, 0, stream>>>(FUSERGB, WB + WB_RGB_PROJ, f32in(P_RGB_PROJ_B),
      XRGB, RES_RGB, nullptr, 256, 0, 0);
  gemm_bf16_kernel<256><<<g4, blk, 0, stream>>>(FUSETH, WB + WB_TH_PROJ, f32in(P_TH_PROJ_B),
      XTH, RES_TH, nullptr, 256, 0, 0);

  // ---- LN2 ----
  ln2_kernel<<<dim3(2 * NPIX / 8), blk, 0, stream>>>(
      RES_RGB, RES_TH,
      f32in(P_RGB_N2_G), f32in(P_RGB_N2_B), f32in(P_TH_N2_G), f32in(P_TH_N2_B),
      LN2_RGB, LN2_TH);

  // ---- MLP fc1 (GELU) ----
  gemm_bf16_kernel<256><<<g8, blk, 0, stream>>>(LN2_RGB, WB + WB_RGB_FC1, nullptr,
      nullptr, nullptr, H_RGB, 512, /*gelu*/2, 0);
  gemm_bf16_kernel<256><<<g8, blk, 0, stream>>>(LN2_TH, WB + WB_TH_FC1, nullptr,
      nullptr, nullptr, H_TH, 512, /*gelu*/2, 0);

  // ---- MLP fc2 + residual, scatter fp32 directly to (B,C,H,W) output ----
  float* out_rgb = (float*)d_out;
  float* out_th  = (float*)d_out + (size_t)NPIX * CH;
  gemm_bf16_kernel<512><<<g4, blk, 0, stream>>>(H_RGB, WB + WB_RGB_FC2, nullptr,
      RES_RGB, out_rgb, nullptr, 256, 0, /*scatter*/1);
  gemm_bf16_kernel<512><<<g4, blk, 0, stream>>>(H_TH, WB + WB_TH_FC2, nullptr,
      RES_TH, out_th, nullptr, 256, 0, /*scatter*/1);
}
